// BitConv2d_80315888435851
// MI455X (gfx1250) — 8-bit matrix operand order repaired and checked by reading, compiled, not run on hardware
//
#include <hip/hip_runtime.h>
#include <stdint.h>

typedef __attribute__((ext_vector_type(8))) int v8i;

#define EPSQ 1e-8f
#define QMAX 127.0f

#define WS_ABSMAX 0
#define WS_SW     256
#define WS_WQ     1024

__global__ void bc_init(unsigned int* absmax) {
    if (threadIdx.x == 0 && blockIdx.x == 0) absmax[0] = 0u;
}

__global__ __launch_bounds__(256) void bc_absmax(const float4* __restrict__ x4,
                                                 unsigned int* __restrict__ absmax,
                                                 int n4) {
    float m = 0.0f;
    int stride = gridDim.x * blockDim.x;
    for (int i = blockIdx.x * blockDim.x + threadIdx.x; i < n4; i += stride) {
        float4 v = x4[i];
        m = fmaxf(m, fminf(fabsf(v.x), 8.0f));
        m = fmaxf(m, fminf(fabsf(v.y), 8.0f));
        m = fmaxf(m, fminf(fabsf(v.z), 8.0f));
        m = fmaxf(m, fminf(fabsf(v.w), 8.0f));
    }
    #pragma unroll
    for (int off = 16; off > 0; off >>= 1)
        m = fmaxf(m, __shfl_xor(m, off, 32));
    if ((threadIdx.x & 31) == 0)
        atomicMax(absmax, __float_as_uint(m));
}

__global__ void bc_wquant(const float* __restrict__ w,
                          float* __restrict__ sw,
                          int8_t* __restrict__ wq) {
    int o = threadIdx.x;
    if (o >= 64) return;
    float s = 0.0f;
    #pragma unroll 4
    for (int i = 0; i < 64; ++i) s += fabsf(w[o * 64 + i]);
    s *= (1.0f / 64.0f);
    s = fmaxf(s, EPSQ);
    sw[o] = s;
    float inv = 1.0f / s;
    #pragma unroll 4
    for (int i = 0; i < 64; ++i) {
        float q = rintf(w[o * 64 + i] * inv);
        q = fminf(fmaxf(q, -1.0f), 1.0f);
        wq[o * 64 + i] = (int8_t)(int)q;
    }
}

__device__ __forceinline__ int bc_quant1(float v, float inv_s) {
    v = fminf(fmaxf(v, -8.0f), 8.0f);
    float q = rintf(v * inv_s);
    q = fminf(fmaxf(q, -QMAX), QMAX);
    return (int)q;
}

__device__ __forceinline__ unsigned int bc_pack4(int q0, int q1, int q2, int q3) {
    unsigned int lo = __builtin_amdgcn_perm((unsigned)q1, (unsigned)q0, 0x0c0c0400u);
    unsigned int hi = __builtin_amdgcn_perm((unsigned)q3, (unsigned)q2, 0x04000c0cu);
    return lo | hi;
}

__global__ __launch_bounds__(256) void bc_gemm(const float* __restrict__ x,
                                               float* __restrict__ out,
                                               const unsigned int* __restrict__ absmax_u,
                                               const float* __restrict__ sw,
                                               const int* __restrict__ wq_dw) {
    const int lane = threadIdx.x & 31;
    const int wave = threadIdx.x >> 5;
    const int g    = lane >> 4;
    const int lm   = lane & 15;

    const float amax  = __uint_as_float(absmax_u[0]);
    const float s_act = fmaxf(amax, EPSQ) * (1.0f / QMAX);
    const float inv_s = 1.0f / s_act;

    v8i A[4];
    #pragma unroll
    for (int t = 0; t < 4; ++t) {
        const int M = t * 16 + lm;
        #pragma unroll
        for (int j = 0; j < 8; ++j) {
            const int K0 = ((j & 1) * 4) + (((j >> 1) & 1) * 16) + ((j >> 2) * 32) + g * 8;
            A[t][j] = wq_dw[M * 16 + (K0 >> 2)];
        }
    }

    float sc[4][8];
    #pragma unroll
    for (int t = 0; t < 4; ++t)
        #pragma unroll
        for (int r = 0; r < 8; ++r)
            sc[t][r] = s_act * sw[t * 16 + r + 8 * g];

    const int waveGlobal = 16383 - (blockIdx.x * 8 + wave);
    const int tileBase   = waveGlobal * 4;

    for (int tt = 0; tt < 4; ++tt) {
        const int tile  = tileBase + tt;
        const int nb    = tile >> 12;
        const int ptile = (tile & 4095) << 4;
        const int n     = ptile + lm;
        const float* xb = x + (size_t)nb * 64 * 65536;

        v8i B;
        #pragma unroll
        for (int j = 0; j < 8; ++j) {
            const int K0 = ((j & 1) * 4) + (((j >> 1) & 1) * 16) + ((j >> 2) * 32) + g * 8;
            const float v0 = __builtin_nontemporal_load(&xb[(size_t)(K0 + 0) * 65536 + n]);
            const float v1 = __builtin_nontemporal_load(&xb[(size_t)(K0 + 1) * 65536 + n]);
            const float v2 = __builtin_nontemporal_load(&xb[(size_t)(K0 + 2) * 65536 + n]);
            const float v3 = __builtin_nontemporal_load(&xb[(size_t)(K0 + 3) * 65536 + n]);
            B[j] = (int)bc_pack4(bc_quant1(v0, inv_s), bc_quant1(v1, inv_s),
                                 bc_quant1(v2, inv_s), bc_quant1(v3, inv_s));
        }

        v8i acc[4];
        #pragma unroll
        for (int t = 0; t < 4; ++t) {
            v8i c = {};
            acc[t] = __builtin_amdgcn_wmma_i32_16x16x64_iu8(
                true, A[t], true, B, c, false, false);
        }

        float* ob = out + (size_t)nb * 64 * 65536 + n;
        #pragma unroll
        for (int t = 0; t < 4; ++t) {
            #pragma unroll
            for (int r = 0; r < 8; ++r) {
                const int o = t * 16 + r + 8 * g;
                __builtin_nontemporal_store((float)acc[t][r] * sc[t][r],
                                            &ob[(size_t)o * 65536]);
            }
        }
    }
}

extern "C" void kernel_launch(void* const* d_in, const int* in_sizes, int n_in,
                              void* d_out, int out_size, void* d_ws, size_t ws_size,
                              hipStream_t stream) {
    const float* x = (const float*)d_in[0];
    const float* w = (const float*)d_in[1];
    float* out     = (float*)d_out;

    char* ws = (char*)d_ws;
    unsigned int* absmax = (unsigned int*)(ws + WS_ABSMAX);
    float*        sw     = (float*)(ws + WS_SW);
    int8_t*       wq     = (int8_t*)(ws + WS_WQ);
    int*          wq_dw  = (int*)(ws + WS_WQ);

    bc_init<<<1, 32, 0, stream>>>(absmax);

    const int n4 = 16 * 64 * 256 * 256 / 4;
    bc_absmax<<<4096, 256, 0, stream>>>((const float4*)x, absmax, n4);

    bc_wquant<<<1, 64, 0, stream>>>(w, sw, wq);

    bc_gemm<<<2048, 256, 0, stream>>>(x, out, absmax, sw, wq_dw);
}
